// DGCNLayer_77111842832925
// MI455X (gfx1250) — matrix operand order repaired and checked by reading, not compiled, not run on hardware
//
#include <hip/hip_runtime.h>

// ---------------------------------------------------------------------------
// CDNA5 (gfx1250) DGCN layer: bf16 WMMA GEMMs + CSR SpMM
// ---------------------------------------------------------------------------

typedef __attribute__((ext_vector_type(16))) __bf16 v16bf;
typedef __attribute__((ext_vector_type(8)))  float  v8f;

#define LRELU_SLOPE 0.2f

// ---- f32 -> bf16 conversion with zero padding of tail rows -----------------
__global__ void cvt_pad_kernel(const float* __restrict__ src, __bf16* __restrict__ dst,
                               int n_src, int n_dst) {
  int i = blockIdx.x * blockDim.x + threadIdx.x;
  if (i >= n_dst) return;
  float v = (i < n_src) ? src[i] : 0.f;
  dst[i] = (__bf16)v;
}

// ---- pack weight W[K,128] (f32 row-major) into WMMA B-fragment layout ------
// Fragment (kc,nt) holds the 32x16 tile B[kc*32 .. kc*32+31][nt*16 .. nt*16+15].
// B layout used here (contiguous k): lane l, elem e  ->  K = (l>>4)*16 + e, N = l&15.
// Stored contiguously: frag*512 + lane*16 + e   (bf16 elements)
__global__ void pack_weight_kernel(const float* __restrict__ W, __bf16* __restrict__ dst) {
  int frag = blockIdx.x;            // kc*8 + nt
  int kc = frag >> 3, nt = frag & 7;
  int l  = threadIdx.x;             // 32 threads = one wave
  int kb = kc * 32 + ((l >> 4) << 4);
  int n  = nt * 16 + (l & 15);
  __bf16* o = dst + (size_t)frag * 512 + (size_t)l * 16;
  #pragma unroll
  for (int e = 0; e < 16; ++e)
    o[e] = (__bf16)W[(size_t)(kb + e) * 128 + n];
}

// ---- CSR row pointers from sorted COO rows (binary search) -----------------
__global__ void build_rowptr_kernel(const int* __restrict__ rows, int E, int n_rows,
                                    int* __restrict__ ptr) {
  int r = blockIdx.x * blockDim.x + threadIdx.x;
  if (r > n_rows) return;
  if (r == n_rows) { ptr[r] = E; return; }
  int lo = 0, hi = E;
  while (lo < hi) { int mid = (lo + hi) >> 1; if (rows[mid] < r) lo = mid + 1; else hi = mid; }
  ptr[r] = lo;
}

// ---- SpMM + bias + LeakyReLU: Y[r,:] = lrelu(sum_e vals[e]*X[cols[e],:] + b)
// one block per row, 128 threads = feature columns; bf16 gathers, f32 accum
__global__ __launch_bounds__(128)
void spmm_bias_lrelu_kernel(const int* __restrict__ rowptr,
                            const int* __restrict__ cols,
                            const float* __restrict__ vals,
                            const __bf16* __restrict__ X,
                            const float* __restrict__ bias,
                            __bf16* __restrict__ Y) {
  int r = blockIdx.x;
  int t = threadIdx.x;
  int s = rowptr[r], e = rowptr[r + 1];
  float acc = 0.f;
  for (int k = s; k < e; ++k) {
    int   c = cols[k];
    float v = vals[k];
    acc = fmaf(v, (float)X[(size_t)c * 128 + t], acc);
  }
  float y = acc + bias[t];
  y = (y > 0.f) ? y : LRELU_SLOPE * y;
  Y[(size_t)r * 128 + t] = (__bf16)y;
}

// ---- WMMA GEMM: C[M,128] = act( [A1 | A2] @ Wp + bias ) --------------------
// KC = 4  : C = A1[M,128] @ W[128,128]            (mode 0: raw bf16 out)
// KC = 8  : C = concat(A1,A2)[M,256] @ W[256,128] (mode 1: f32 relu+bias out)
// 256 threads = 8 waves; each wave computes a 16x128 strip via 8 WMMA tiles.
// A fragment (16x32, contiguous k as in packed B): lane l, elem e:
//   M = l&15,  K = (l>>4)*16 + e
union FragCast { v16bf v; uint4 q[2]; };

template <int KC, int MODE>
__global__ __launch_bounds__(256)
void gemm_bf16_wmma_kernel(const __bf16* __restrict__ A1, const __bf16* __restrict__ A2,
                           const __bf16* __restrict__ Wp, const float* __restrict__ bias,
                           void* __restrict__ out, int M) {
  const int lane = threadIdx.x & 31;
  const int wave = threadIdx.x >> 5;
  const int half = lane >> 4;
  const int l16  = lane & 15;
  const int rowbase = blockIdx.x * 128 + wave * 16;
  const int arow = rowbase + l16;

  v8f acc[8] = {};

  #pragma unroll
  for (int kc = 0; kc < KC; ++kc) {
    const __bf16* A = (KC == 8 && kc >= 4) ? A2 : A1;
    const int kb = (kc & 3) * 32 + half * 16;
    const __bf16* ap = A + (size_t)arow * 128 + kb;
    FragCast af;
    af.q[0] = *(const uint4*)(ap);        // elems 0..7  : K = kb .. kb+7
    af.q[1] = *(const uint4*)(ap + 8);    // elems 8..15 : K = kb+8 .. kb+15
    const __bf16* wb = Wp + (size_t)(kc * 8) * 512 + (size_t)lane * 16;
    #pragma unroll
    for (int nt = 0; nt < 8; ++nt) {
      const __bf16* bp = wb + (size_t)nt * 512;
      FragCast bfr;
      bfr.q[0] = *(const uint4*)(bp);
      bfr.q[1] = *(const uint4*)(bp + 8);
      acc[nt] = __builtin_amdgcn_wmma_f32_16x16x32_bf16(
          false, af.v, false, bfr.v, (short)0, acc[nt], false, false);
    }
  }

  // C/D layout: reg i -> row (lane>>4)*8 + i of the 16x16 tile, col = lane&15
  if (MODE == 0) {
    __bf16* O = (__bf16*)out;
    #pragma unroll
    for (int nt = 0; nt < 8; ++nt) {
      #pragma unroll
      for (int i = 0; i < 8; ++i) {
        int r = rowbase + half * 8 + i;
        if (r < M) O[(size_t)r * 128 + nt * 16 + l16] = (__bf16)acc[nt][i];
      }
    }
  } else {
    float* O = (float*)out;
    #pragma unroll
    for (int nt = 0; nt < 8; ++nt) {
      float b = bias[nt * 16 + l16];
      #pragma unroll
      for (int i = 0; i < 8; ++i) {
        int r = rowbase + half * 8 + i;
        float v = acc[nt][i] + b;
        v = (v > 0.f) ? v : 0.f;
        if (r < M) O[(size_t)r * 128 + nt * 16 + l16] = v;
      }
    }
  }
}

// ---------------------------------------------------------------------------
extern "C" void kernel_launch(void* const* d_in, const int* in_sizes, int n_in,
                              void* d_out, int out_size, void* d_ws, size_t ws_size,
                              hipStream_t stream) {
  const float* ufea   = (const float*)d_in[0];
  const float* vfea   = (const float*)d_in[1];
  const int*   uvRows = (const int*)  d_in[2];
  const int*   uvCols = (const int*)  d_in[3];
  const float* uvVals = (const float*)d_in[4];
  const int*   vuRows = (const int*)  d_in[5];
  const int*   vuCols = (const int*)  d_in[6];
  const float* vuVals = (const float*)d_in[7];
  const float* W1 = (const float*)d_in[8];  const float* b1 = (const float*)d_in[9];
  const float* W2 = (const float*)d_in[10]; const float* b2 = (const float*)d_in[11];
  const float* W3 = (const float*)d_in[12]; const float* b3 = (const float*)d_in[13];
  const float* W4 = (const float*)d_in[14]; const float* b4 = (const float*)d_in[15];
  const float* Wu = (const float*)d_in[16]; const float* bu = (const float*)d_in[17];
  const float* Wi = (const float*)d_in[18]; const float* bi = (const float*)d_in[19];

  const int F   = 128;
  const int U   = in_sizes[0] / F;
  const int V   = in_sizes[1] / F;
  const int Euv = in_sizes[2];
  const int Evu = in_sizes[5];
  const int Upad = ((U + 127) / 128) * 128;
  const int Vpad = ((V + 127) / 128) * 128;
  const int NP   = (Upad > Vpad) ? Upad : Vpad;

  // ---- carve workspace (all regions 256B aligned) ----
  char* w = (char*)d_ws;
  auto take = [&](size_t bytes) -> char* {
    char* p = w; w += (bytes + 255) & ~(size_t)255; return p;
  };
  __bf16* ufea_bf = (__bf16*)take((size_t)Upad * F * 2);
  __bf16* vfea_bf = (__bf16*)take((size_t)Vpad * F * 2);
  __bf16* P       = (__bf16*)take((size_t)NP   * F * 2);   // GEMM product buffer
  __bf16* userHo  = (__bf16*)take((size_t)Vpad * F * 2);   // [V,H]
  __bf16* itemHo  = (__bf16*)take((size_t)Upad * F * 2);   // [U,H]
  __bf16* uHo2    = (__bf16*)take((size_t)Upad * F * 2);   // [U,F]
  __bf16* iHo2    = (__bf16*)take((size_t)Vpad * F * 2);   // [V,F]
  __bf16* W1p = (__bf16*)take(32 * 512 * 2);
  __bf16* W2p = (__bf16*)take(32 * 512 * 2);
  __bf16* W3p = (__bf16*)take(32 * 512 * 2);
  __bf16* W4p = (__bf16*)take(32 * 512 * 2);
  __bf16* Wup = (__bf16*)take(64 * 512 * 2);
  __bf16* Wip = (__bf16*)take(64 * 512 * 2);
  int* uvPtr = (int*)take((size_t)(U + 1) * 4);
  int* vuPtr = (int*)take((size_t)(V + 1) * 4);

  // ---- prep: bf16 features, packed weights, CSR row pointers ----
  {
    int n = Upad * F;
    cvt_pad_kernel<<<(n + 255) / 256, 256, 0, stream>>>(ufea, ufea_bf, U * F, n);
    n = Vpad * F;
    cvt_pad_kernel<<<(n + 255) / 256, 256, 0, stream>>>(vfea, vfea_bf, V * F, n);
  }
  pack_weight_kernel<<<32, 32, 0, stream>>>(W1, W1p);
  pack_weight_kernel<<<32, 32, 0, stream>>>(W2, W2p);
  pack_weight_kernel<<<32, 32, 0, stream>>>(W3, W3p);
  pack_weight_kernel<<<32, 32, 0, stream>>>(W4, W4p);
  pack_weight_kernel<<<64, 32, 0, stream>>>(Wu, Wup);
  pack_weight_kernel<<<64, 32, 0, stream>>>(Wi, Wip);
  build_rowptr_kernel<<<(U + 1 + 255) / 256, 256, 0, stream>>>(uvRows, Euv, U, uvPtr);
  build_rowptr_kernel<<<(V + 1 + 255) / 256, 256, 0, stream>>>(vuRows, Evu, V, vuPtr);

  const int gU = Upad / 128, gV = Vpad / 128;

  // ---- GCN layer 1 ----
  // user_ho = lrelu(spmm(VU, ufea@W1) + b1)   [V,H]
  gemm_bf16_wmma_kernel<4, 0><<<gU, 256, 0, stream>>>(ufea_bf, ufea_bf, W1p, nullptr, P, U);
  spmm_bias_lrelu_kernel<<<V, 128, 0, stream>>>(vuPtr, vuCols, vuVals, P, b1, userHo);
  // item_ho = lrelu(spmm(UV, vfea@W2) + b2)   [U,H]
  gemm_bf16_wmma_kernel<4, 0><<<gV, 256, 0, stream>>>(vfea_bf, vfea_bf, W2p, nullptr, P, V);
  spmm_bias_lrelu_kernel<<<U, 128, 0, stream>>>(uvPtr, uvCols, uvVals, P, b2, itemHo);

  // ---- GCN layer 2 ----
  // user_ho2 = lrelu(spmm(UV, user_ho@W3) + b3)  [U,F]
  gemm_bf16_wmma_kernel<4, 0><<<gV, 256, 0, stream>>>(userHo, userHo, W3p, nullptr, P, V);
  spmm_bias_lrelu_kernel<<<U, 128, 0, stream>>>(uvPtr, uvCols, uvVals, P, b3, uHo2);
  // item_ho2 = lrelu(spmm(VU, item_ho@W4) + b4)  [V,F]
  gemm_bf16_wmma_kernel<4, 0><<<gU, 256, 0, stream>>>(itemHo, itemHo, W4p, nullptr, P, U);
  spmm_bias_lrelu_kernel<<<V, 128, 0, stream>>>(vuPtr, vuCols, vuVals, P, b4, iHo2);

  // ---- union MLPs (concat folded into 2-pointer K=256 GEMM, fused bias+ReLU)
  float* outUser = (float*)d_out;
  float* outItem = outUser + (size_t)U * F;
  gemm_bf16_wmma_kernel<8, 1><<<gU, 256, 0, stream>>>(uHo2, ufea_bf, Wup, bu, outUser, U);
  gemm_bf16_wmma_kernel<8, 1><<<gV, 256, 0, stream>>>(iHo2, vfea_bf, Wip, bi, outItem, V);
}
